// GCN_88639535055109
// MI455X (gfx1250) — matrix operand order repaired and checked by reading, not compiled, not run on hardware
//
#include <hip/hip_runtime.h>
#include <hip/hip_bf16.h>

typedef __attribute__((ext_vector_type(16))) _Float16 v16h;
typedef __attribute__((ext_vector_type(8)))  float    v8f;
typedef __attribute__((ext_vector_type(4)))  _Float16 v4h;
typedef __attribute__((ext_vector_type(4)))  float    v4f;

#define NN   100000      // nodes
#define NE   1000000     // edges
#define FD   64          // feature dim (IN_F == H_F)
#define NCLS 10
#define NG   64
#define NT   (NN / 16)   // 6250 row tiles of 16

// ---------------------------------------------------------------- utilities
__global__ void k_zero(float* __restrict__ p, int n) {
    int i = blockIdx.x * blockDim.x + threadIdx.x;
    if (i < n) p[i] = 0.0f;
}

// ------------------------------------------------------------------ degrees
__global__ void k_deg(const int* __restrict__ src, const int* __restrict__ dst,
                      float* __restrict__ odeg, float* __restrict__ ideg) {
    int e = blockIdx.x * blockDim.x + threadIdx.x;
    if (e < NE) {
        atomicAdd(&odeg[src[e]], 1.0f);
        atomicAdd(&ideg[dst[e]], 1.0f);
    }
}

// ------------------------------------------- norms + per-graph node counts
__global__ void k_norm(const float* __restrict__ odeg, const float* __restrict__ ideg,
                       const int* __restrict__ gid,
                       float* __restrict__ onorm, float* __restrict__ inorm,
                       float* __restrict__ counts) {
    int n = blockIdx.x * blockDim.x + threadIdx.x;
    if (n < NN) {
        onorm[n] = rsqrtf(fmaxf(odeg[n], 1.0f));
        inorm[n] = rsqrtf(fmaxf(ideg[n], 1.0f));
        atomicAdd(&counts[gid[n]], 1.0f);
    }
}

// ----------------------- layer-1 message prep: xs = f16(in_feat * out_norm)
__global__ void k_scale(const float* __restrict__ x, const float* __restrict__ onorm,
                        _Float16* __restrict__ xs) {
    int t = blockIdx.x * blockDim.x + threadIdx.x;
    if (t < NN * 16) {
        int n = t >> 4, q = t & 15;
        float s = onorm[n];
        v4f v = *(const v4f*)(x + (size_t)n * FD + q * 4);
        v4h h;
        h[0] = (_Float16)(v[0] * s); h[1] = (_Float16)(v[1] * s);
        h[2] = (_Float16)(v[2] * s); h[3] = (_Float16)(v[3] * s);
        *(v4h*)(xs + (size_t)n * FD + q * 4) = h;
    }
}

// --------------------- transpose weights to f16 column-major (W2 padded->16)
__global__ void k_prepw(const float* __restrict__ W1, const float* __restrict__ W2,
                        _Float16* __restrict__ W1T, _Float16* __restrict__ W2T) {
    int t = blockIdx.x * blockDim.x + threadIdx.x;
    if (t < 64 * 64) {
        int c = t >> 6, k = t & 63;
        W1T[c * 64 + k] = (_Float16)W1[k * 64 + c];
    } else if (t < 64 * 64 + 16 * 64) {
        int u = t - 64 * 64;
        int c = u >> 6, k = u & 63;
        W2T[c * 64 + k] = (c < NCLS) ? (_Float16)W2[k * NCLS + c] : (_Float16)0.0f;
    }
}

// ----------------------------- edge gather/scatter: agg[dst] += f32(xs[src])
__global__ void k_scatter(const int* __restrict__ src, const int* __restrict__ dst,
                          const _Float16* __restrict__ xs, float* __restrict__ agg) {
    int t = blockIdx.x * blockDim.x + threadIdx.x;   // NE*16 = 16M items
    if (t < NE * 16) {
        int e = t >> 4, q = t & 15;
        int s = src[e], d = dst[e];
        v4h h = *(const v4h*)(xs + (size_t)s * FD + q * 4);
        float* out = agg + (size_t)d * FD + q * 4;
        atomicAdd(out + 0, (float)h[0]);
        atomicAdd(out + 1, (float)h[1]);
        atomicAdd(out + 2, (float)h[2]);
        atomicAdd(out + 3, (float)h[3]);
    }
}

// Build 16x32 f16 A-fragment from agg rows scaled by in_norm (contiguous k = 16*kh + s)
__device__ inline v16h load_a_frag(const float* __restrict__ agg, int rowA, int kh,
                                   int kc, float sc) {
    const float* ap = agg + (size_t)rowA * FD + kc * 32 + kh * 16;
    v4f a0 = *(const v4f*)(ap);       v4f a1 = *(const v4f*)(ap + 4);
    v4f a2 = *(const v4f*)(ap + 8);   v4f a3 = *(const v4f*)(ap + 12);
    v16h a;
#pragma unroll
    for (int j = 0; j < 4; ++j) {
        a[j]      = (_Float16)(a0[j] * sc);
        a[4 + j]  = (_Float16)(a1[j] * sc);
        a[8 + j]  = (_Float16)(a2[j] * sc);
        a[12 + j] = (_Float16)(a3[j] * sc);
    }
    return a;
}

// -------- layer-1 GEMM: xs2 = f16( relu( (agg*in_norm) @ W1 + b1 ) * out_norm )
__global__ void __launch_bounds__(256)
k_gemm1(const float* __restrict__ agg, const float* __restrict__ inorm,
        const float* __restrict__ onorm, const _Float16* __restrict__ W1T,
        const float* __restrict__ b1, _Float16* __restrict__ xs2) {
    int wave = threadIdx.x >> 5, lane = threadIdx.x & 31;
    int tile = blockIdx.x * 8 + wave;
    if (tile >= NT) return;                       // wave-uniform: EXEC stays full
    int kh   = lane >> 4;
    int rowA = tile * 16 + (lane & 15);
    float sc = inorm[rowA];

    v8f acc[4] = {};
#pragma unroll
    for (int kc = 0; kc < 2; ++kc) {
        v16h a = load_a_frag(agg, rowA, kh, kc, sc);
        int kb = kc * 32 + kh * 16;
#pragma unroll
        for (int nb = 0; nb < 4; ++nb) {
            v16h b = *(const v16h*)(W1T + (size_t)(nb * 16 + (lane & 15)) * 64 + kb);
            acc[nb] = __builtin_amdgcn_wmma_f32_16x16x32_f16(
                false, a, false, b, (short)0, acc[nb], false, false);
        }
    }
    int col = lane & 15;
#pragma unroll
    for (int r = 0; r < 8; ++r) {
        int row = tile * 16 + r + 8 * kh;
        float on = onorm[row];
#pragma unroll
        for (int nb = 0; nb < 4; ++nb) {
            int c = nb * 16 + col;
            float v = fmaxf(acc[nb][r] + b1[c], 0.0f) * on;
            xs2[(size_t)row * FD + c] = (_Float16)v;
        }
    }
}

// -------- layer-2 GEMM + per-graph sum: sums[g] += (agg*in_norm) @ W2 + b2
__global__ void __launch_bounds__(256)
k_gemm2(const float* __restrict__ agg, const float* __restrict__ inorm,
        const _Float16* __restrict__ W2T, const float* __restrict__ b2,
        const int* __restrict__ gid, float* __restrict__ sums) {
    int wave = threadIdx.x >> 5, lane = threadIdx.x & 31;
    int tile = blockIdx.x * 8 + wave;
    if (tile >= NT) return;
    int kh   = lane >> 4;
    int rowA = tile * 16 + (lane & 15);
    float sc = inorm[rowA];
    int col  = lane & 15;

    v8f acc = {};
#pragma unroll
    for (int kc = 0; kc < 2; ++kc) {
        v16h a = load_a_frag(agg, rowA, kh, kc, sc);
        v16h b = *(const v16h*)(W2T + (size_t)col * 64 + kc * 32 + kh * 16);
        acc = __builtin_amdgcn_wmma_f32_16x16x32_f16(
            false, a, false, b, (short)0, acc, false, false);
    }
    if (col < NCLS) {                 // divergence only after all WMMAs
        float bias = b2[col];
#pragma unroll
        for (int r = 0; r < 8; ++r) {
            int row = tile * 16 + r + 8 * kh;
            atomicAdd(&sums[gid[row] * NCLS + col], acc[r] + bias);
        }
    }
}

// ------------------------------------------------------------- final mean
__global__ void k_final(const float* __restrict__ sums, const float* __restrict__ counts,
                        float* __restrict__ out) {
    int t = blockIdx.x * blockDim.x + threadIdx.x;
    if (t < NG * NCLS) {
        int g = t / NCLS;
        out[t] = sums[t] / fmaxf(counts[g], 1.0f);
    }
}

// =========================================================================
extern "C" void kernel_launch(void* const* d_in, const int* in_sizes, int n_in,
                              void* d_out, int out_size, void* d_ws, size_t ws_size,
                              hipStream_t stream) {
    const float* in_feat = (const float*)d_in[0];
    const int*   src     = (const int*)d_in[1];
    const int*   dst     = (const int*)d_in[2];
    const int*   gid     = (const int*)d_in[3];
    const float* W1      = (const float*)d_in[4];
    const float* b1      = (const float*)d_in[5];
    const float* W2      = (const float*)d_in[6];
    const float* b2      = (const float*)d_in[7];
    float* out = (float*)d_out;

    // ---- workspace layout (256B-aligned partitions) ----
    char* w = (char*)d_ws;
    size_t off = 0;
    auto alloc = [&](size_t bytes) { void* p = w + off;
                                     off = (off + bytes + 255) & ~(size_t)255; return p; };
    float*    odeg   = (float*)alloc(sizeof(float) * NN);
    float*    ideg   = (float*)alloc(sizeof(float) * NN);
    float*    onorm  = (float*)alloc(sizeof(float) * NN);
    float*    inorm  = (float*)alloc(sizeof(float) * NN);
    float*    counts = (float*)alloc(sizeof(float) * NG);
    float*    sums   = (float*)alloc(sizeof(float) * NG * NCLS);
    _Float16* W1T    = (_Float16*)alloc(sizeof(_Float16) * 64 * 64);
    _Float16* W2T    = (_Float16*)alloc(sizeof(_Float16) * 16 * 64);
    float*    agg    = (float*)alloc(sizeof(float) * (size_t)NN * FD);   // shared L1/L2
    _Float16* xs     = (_Float16*)alloc(sizeof(_Float16) * (size_t)NN * FD); // shared

    const int T = 256;
    auto blk = [](long n, int t) { return (int)((n + t - 1) / t); };

    // zero accumulators
    k_zero<<<blk(NN, T), T, 0, stream>>>(odeg, NN);
    k_zero<<<blk(NN, T), T, 0, stream>>>(ideg, NN);
    k_zero<<<blk((long)NN * FD, T), T, 0, stream>>>(agg, NN * FD);
    k_zero<<<blk(NG * NCLS, T), T, 0, stream>>>(sums, NG * NCLS);
    k_zero<<<1, 64, 0, stream>>>(counts, NG);

    k_prepw<<<blk(64 * 64 + 16 * 64, T), T, 0, stream>>>(W1, W2, W1T, W2T);
    k_deg<<<blk(NE, T), T, 0, stream>>>(src, dst, odeg, ideg);
    k_norm<<<blk(NN, T), T, 0, stream>>>(odeg, ideg, gid, onorm, inorm, counts);

    // ---- layer 1 ----
    k_scale<<<blk((long)NN * 16, T), T, 0, stream>>>(in_feat, onorm, xs);
    k_scatter<<<blk((long)NE * 16, T), T, 0, stream>>>(src, dst, xs, agg);
    k_gemm1<<<blk(NT, 8), T, 0, stream>>>(agg, inorm, onorm, W1T, b1, xs);

    // ---- layer 2 (reuse agg buffer) ----
    k_zero<<<blk((long)NN * FD, T), T, 0, stream>>>(agg, NN * FD);
    k_scatter<<<blk((long)NE * 16, T), T, 0, stream>>>(src, dst, xs, agg);
    k_gemm2<<<blk(NT, 8), T, 0, stream>>>(agg, inorm, W2T, b2, gid, sums);

    // ---- graph mean ----
    k_final<<<blk(NG * NCLS, T), T, 0, stream>>>(sums, counts, out);
}
